// STFT_57578331571000
// MI455X (gfx1250) — matrix operand order repaired and checked by reading, not compiled, not run on hardware
//
#include <hip/hip_runtime.h>
#include <hip/hip_bf16.h>
#include <stdint.h>

// ---------------- problem constants ----------------
#define SIG_L   220500
#define BATCH   32
#define NFFT    2048
#define HOP     512
#define PADW    1024
#define BINS    1025
#define NFRAMES 431                      // (220500/512)+1
#define M_TOT   (BATCH * NFRAMES)        // 13792
#define N_TOT   (2 * BINS)               // 2050 (interleaved: even=cos, odd=-sin)
#define K_TOT   NFFT                     // 2048
#define KSTEP   32
#define KSTEPS  (K_TOT / KSTEP)          // 64
#define BLK_M   128
#define BLK_N   128
#define MBLKS   ((M_TOT + BLK_M - 1) / BLK_M)   // 108
#define NBLKS   ((N_TOT + BLK_N - 1) / BLK_N)   // 17
#define NT_TILES (NBLKS * 8)             // 136 padded N subtiles
#define MT_TILES (MBLKS * 8)             // 864 padded M subtiles

typedef __bf16  bf16_t;
typedef bf16_t  v16bf __attribute__((ext_vector_type(16)));
typedef float   v8f   __attribute__((ext_vector_type(8)));

// workspace layout (bytes)
#define WF_BYTES ((size_t)NT_TILES * KSTEPS * 32 * 32)   //  ~8.9 MB
#define AF_BYTES ((size_t)MT_TILES * KSTEPS * 32 * 32)   // ~56.6 MB

// ---------------------------------------------------------------------------
// Prep 1: pack interleaved weights f32 -> bf16 B-fragments.
// Row n (0..2049): bin = n>>1 ; n even -> wcos[bin], n odd -> -wsin[bin].
// slot s = ((nt*KSTEPS)+ks)*32 + lane ; 16 bf16 per slot.
// B fragment (16-bit, 32x16): lane L = column N=L%16, K = (L/16)*16 + j.
// ---------------------------------------------------------------------------
__global__ __launch_bounds__(256) void stft_pack_weights(
    const float* __restrict__ wcos, const float* __restrict__ wsin,
    bf16_t* __restrict__ wf)
{
    int s = blockIdx.x * 256 + threadIdx.x;
    if (s >= NT_TILES * KSTEPS * 32) return;
    int lane = s & 31;
    int ks   = (s >> 5) & (KSTEPS - 1);
    int nt   = s >> 11;
    int n     = nt * 16 + (lane & 15);
    int kbase = ks * KSTEP + (lane >> 4) * 16;
    int bin   = n >> 1;

    v16bf t;
#pragma unroll
    for (int j = 0; j < 16; ++j) {
        int k = kbase + j;
        float v = 0.0f;
        if (n < N_TOT)
            v = (n & 1) ? -wsin[(size_t)bin * K_TOT + k]
                        :  wcos[(size_t)bin * K_TOT + k];
        t[j] = (bf16_t)v;
    }
    *(v16bf*)(wf + (size_t)s * 16) = t;
}

// 8 contiguous window samples with reflect padding; fast path when inside.
__device__ __forceinline__ void load8_reflect(const float* __restrict__ xb,
                                              int p0, float* v)
{
    if (p0 >= 0 && p0 + 8 <= SIG_L) {
        const float4* q = (const float4*)(xb + p0);
        float4 a = q[0], b = q[1];
        v[0]=a.x; v[1]=a.y; v[2]=a.z; v[3]=a.w;
        v[4]=b.x; v[5]=b.y; v[6]=b.z; v[7]=b.w;
    } else {
#pragma unroll
        for (int j = 0; j < 8; ++j) {
            int p = p0 + j;
            p = (p < 0) ? -p : p;
            p = (p >= SIG_L) ? (2 * SIG_L - 2 - p) : p;
            v[j] = xb[p];
        }
    }
}

// ---------------------------------------------------------------------------
// Prep 2: pack frames (reflect-padded, f32 -> bf16) into A-fragments ONCE.
// slot s = ((mt*KSTEPS)+ks)*32 + lane.
// A fragment (16-bit, 16x32): lane L = row M=L%16, h=L/16;
//   values 0..15 -> K = h*16 + j.
// ---------------------------------------------------------------------------
__global__ __launch_bounds__(256) void stft_pack_frames(
    const float* __restrict__ x, bf16_t* __restrict__ af)
{
    int s = blockIdx.x * 256 + threadIdx.x;
    if (s >= MT_TILES * KSTEPS * 32) return;
    int lane = s & 31;
    int ks   = (s >> 5) & (KSTEPS - 1);
    int mt   = s >> 11;

    int m = mt * 16 + (lane & 15);
    if (m >= M_TOT) m = M_TOT - 1;        // padded rows: harmless duplicates
    int bb = m / NFRAMES;
    int tt = m - bb * NFRAMES;
    const float* xb = x + (size_t)bb * SIG_L;
    int h  = lane >> 4;
    int kb = ks * KSTEP + h * 16;

    float v[16];
    load8_reflect(xb, tt * HOP + kb - PADW,      v);
    load8_reflect(xb, tt * HOP + kb + 8 - PADW,  v + 8);

    v16bf t;
#pragma unroll
    for (int j = 0; j < 16; ++j) t[j] = (bf16_t)v[j];
    *(v16bf*)(af + (size_t)s * 16) = t;
}

// ---------------------------------------------------------------------------
// Main GEMM: 8 waves/WG, wave grid 4(M) x 2(N), wave tile 32M x 64N.
// A and B staged with global_load_async_to_lds_b128, TRIPLE-buffered:
// one slab always in flight (s_wait_asynccnt 4); drain peeled out of loop.
// ---------------------------------------------------------------------------
__global__ __launch_bounds__(256) void stft_wmma(
    const bf16_t* __restrict__ af, const bf16_t* __restrict__ wf,
    float* __restrict__ out)
{
    __shared__ v16bf Alds[3][8][32];   // 24KB
    __shared__ v16bf Blds[3][8][32];   // 24KB

    const int tid    = threadIdx.x;
    const int lane   = tid & 31;
    const int wv     = tid >> 5;       // 0..7
    const int waveM  = wv & 3;
    const int waveN  = wv >> 2;
    const int blockM = blockIdx.x;     // 0..107
    const int blockN = blockIdx.y;     // 0..16

    // staging: thread (wv, lane) moves A subtile wv and B subtile wv
    const int mtg = blockM * 8 + wv;   // < MT_TILES
    const int ntg = blockN * 8 + wv;   // < NT_TILES
    const uint64_t gA0 = (uint64_t)(const void*)af +
                         ((uint64_t)mtg * KSTEPS * 32ull + lane) * 32ull;
    const uint64_t gB0 = (uint64_t)(const void*)wf +
                         ((uint64_t)ntg * KSTEPS * 32ull + lane) * 32ull;
    const unsigned ldsAb = (unsigned)(uint64_t)&Alds[0][wv][lane];
    const unsigned ldsBb = (unsigned)(uint64_t)&Blds[0][wv][lane];
    const unsigned BUFSTRIDE = 8 * 32 * sizeof(v16bf);   // 8192 B per buffer

    auto stage = [&](int buf, int ks) {
        uint64_t gaA = gA0 + (uint64_t)ks * 1024ull;   // 32 lanes * 32B
        uint64_t gaB = gB0 + (uint64_t)ks * 1024ull;
        unsigned la  = ldsAb + (unsigned)buf * BUFSTRIDE;
        unsigned lb  = ldsBb + (unsigned)buf * BUFSTRIDE;
        asm volatile(
            "global_load_async_to_lds_b128 %0, %1, off\n\t"
            "global_load_async_to_lds_b128 %0, %1, off offset:16\n\t"
            "global_load_async_to_lds_b128 %2, %3, off\n\t"
            "global_load_async_to_lds_b128 %2, %3, off offset:16"
            :: "v"(la), "v"(gaA), "v"(lb), "v"(gaB) : "memory");
    };

    v8f acc[2][4] = {};

    // i-major WMMA order (all B loads hoisted,
    // single mid-stream dscnt wait for the a1 reload).
    auto compute = [&](int buf) {
        v16bf a0 = Alds[buf][waveM * 2 + 0][lane];
        v16bf a1 = Alds[buf][waveM * 2 + 1][lane];
        v16bf b0 = Blds[buf][waveN * 4 + 0][lane];
        v16bf b1 = Blds[buf][waveN * 4 + 1][lane];
        v16bf b2 = Blds[buf][waveN * 4 + 2][lane];
        v16bf b3 = Blds[buf][waveN * 4 + 3][lane];
        acc[0][0] = __builtin_amdgcn_wmma_f32_16x16x32_bf16(false, a0, false, b0, (short)0, acc[0][0], false, false);
        acc[0][1] = __builtin_amdgcn_wmma_f32_16x16x32_bf16(false, a0, false, b1, (short)0, acc[0][1], false, false);
        acc[0][2] = __builtin_amdgcn_wmma_f32_16x16x32_bf16(false, a0, false, b2, (short)0, acc[0][2], false, false);
        acc[0][3] = __builtin_amdgcn_wmma_f32_16x16x32_bf16(false, a0, false, b3, (short)0, acc[0][3], false, false);
        acc[1][0] = __builtin_amdgcn_wmma_f32_16x16x32_bf16(false, a1, false, b0, (short)0, acc[1][0], false, false);
        acc[1][1] = __builtin_amdgcn_wmma_f32_16x16x32_bf16(false, a1, false, b1, (short)0, acc[1][1], false, false);
        acc[1][2] = __builtin_amdgcn_wmma_f32_16x16x32_bf16(false, a1, false, b2, (short)0, acc[1][2], false, false);
        acc[1][3] = __builtin_amdgcn_wmma_f32_16x16x32_bf16(false, a1, false, b3, (short)0, acc[1][3], false, false);
    };

    stage(0, 0);
    stage(1, 1);
    int cur = 0, nxt = 2;

    // steady state: unconditional wait<=4 keeps one slab always in flight
#pragma unroll 1
    for (int ks = 0; ks < KSTEPS - 2; ++ks) {
        asm volatile("s_wait_asynccnt 0x4" ::: "memory");  // slab ks complete
        __syncthreads();
        stage(nxt, ks + 2);                                // refill oldest buf
        compute(cur);
        cur = (cur == 2) ? 0 : cur + 1;
        nxt = (nxt == 2) ? 0 : nxt + 1;
    }

    // tail: one full drain + one barrier makes BOTH remaining slabs visible
    asm volatile("s_wait_asynccnt 0x0" ::: "memory");
    __syncthreads();
    compute(cur);
    compute((cur == 2) ? 0 : cur + 1);

    // ---- epilogue: interleaved N -> out[b][bin][t][c]; one divide per
    // M-subtile + carry across the 431-frame boundary (r < 8 << 431).
#pragma unroll
    for (int i = 0; i < 2; ++i) {
        int mBase = blockM * BLK_M + (waveM * 2 + i) * 16 + 8 * (lane >> 4);
        int b0i = mBase / NFRAMES;
        int t0i = mBase - b0i * NFRAMES;
#pragma unroll
        for (int j = 0; j < 4; ++j) {
            int nG = blockN * BLK_N + (waveN * 4 + j) * 16 + (lane & 15);
            if (nG >= N_TOT) continue;
            int bin = nG >> 1;
            int c   = nG & 1;
#pragma unroll
            for (int r = 0; r < 8; ++r) {
                if (mBase + r >= M_TOT) continue;
                int t2 = t0i + r, b2 = b0i;
                if (t2 >= NFRAMES) { t2 -= NFRAMES; b2 += 1; }
                size_t oi = (((size_t)b2 * BINS + bin) * NFRAMES + t2) * 2 + c;
                out[oi] = acc[i][j][r];
            }
        }
    }
}

// ---------------------------------------------------------------------------
extern "C" void kernel_launch(void* const* d_in, const int* in_sizes, int n_in,
                              void* d_out, int out_size, void* d_ws, size_t ws_size,
                              hipStream_t stream)
{
    (void)in_sizes; (void)n_in; (void)out_size; (void)ws_size;
    const float* x    = (const float*)d_in[0];
    const float* wcos = (const float*)d_in[1];
    const float* wsin = (const float*)d_in[2];
    float*  out = (float*)d_out;
    bf16_t* wf  = (bf16_t*)d_ws;                               //  8.9 MB
    bf16_t* af  = (bf16_t*)((char*)d_ws + WF_BYTES);           // 56.6 MB

    const int wslots = NT_TILES * KSTEPS * 32;
    stft_pack_weights<<<(wslots + 255) / 256, 256, 0, stream>>>(wcos, wsin, wf);

    const int aslots = MT_TILES * KSTEPS * 32;
    stft_pack_frames<<<(aslots + 255) / 256, 256, 0, stream>>>(x, af);

    dim3 grid(MBLKS, NBLKS);
    stft_wmma<<<grid, 256, 0, stream>>>(af, wf, out);
}
